// LaterallyConnectedLayer_57870389346650
// MI455X (gfx1250) — matrix operand order repaired and checked by reading, not compiled, not run on hardware
//
#include <hip/hip_runtime.h>

typedef __attribute__((ext_vector_type(16))) _Float16 v16h;
typedef __attribute__((ext_vector_type(8)))  _Float16 v8h;
typedef __attribute__((ext_vector_type(8)))  float    v8f;

constexpr float ETA_   = 0.1f;
constexpr float ALPHA_ = 0.1f;
constexpr float THETA_ = 0.1f;
constexpr float EPS_   = 1e-10f;
constexpr float INV_FM = 1.0f / 64.0f;

// Zero-padded channel-first activation plane: (B,128,PH,PW) f16,
// data at [h+2][w+8]; row stride 80 halfs = 160B so every fragment
// group (…*80 + 8 + wc*32 + {0,8} + {0,16} halfs) is 16B aligned.
constexpr int PW = 80, PH = 52, PLANE = PH * PW;   // 4160

static __device__ __forceinline__ int rmap(int i) {   // symmetric_padding(.,2) gather on 48
    return i < 2 ? i + 44 : (i >= 46 ? i - 44 : i);
}
static __device__ __forceinline__ int qmap(int i) {   // pad_activations gather: 52 -> 48
    return i < 4 ? i + 42 : (i >= 48 ? i - 46 : i - 2);
}
static __device__ __forceinline__ v16h cat16(v8h a, v8h b) {
    return __builtin_shufflevector(a, b, 0,1,2,3,4,5,6,7,8,9,10,11,12,13,14,15);
}

// ---------------------------------------------------------------------------
// Stage 1: zero-padded Aa = clip(tile(sympad(A)) + eta*noise, 0), f16.
// ---------------------------------------------------------------------------
__global__ void lcl_build_azp(const float* __restrict__ A, const float* __restrict__ noise,
                              _Float16* __restrict__ Azp) {
    const int idx = blockIdx.x * 256 + threadIdx.x;           // 16*128*PLANE = 8519680
    const int w = idx % PW;
    const int h = (idx / PW) % PH;
    const int c = (idx / PLANE) % 128;
    const int b = idx / (PLANE * 128);
    float r = 0.0f;
    if (h >= 2 && h < 50 && w >= 8 && w < 56) {
        const int hh = h - 2, ww = w - 8;
        const float v = A[(((size_t)b * 64 + (c & 63)) * 48 + rmap(hh)) * 48 + rmap(ww)]
                      + ETA_ * noise[(((size_t)b * 128 + c) * 48 + hh) * 48 + ww];
        r = v > 0.0f ? v : 0.0f;
    }
    Azp[idx] = (_Float16)r;
}

// ---------------------------------------------------------------------------
// Stage 2: A_sum[b,c] = sum(Aa[b,c]) + 1  (zero border doesn't change the sum;
// softmax(L1) map sums to exactly 1, so the whole L1 conv is dead code).
// ---------------------------------------------------------------------------
__global__ void lcl_asum(const _Float16* __restrict__ Azp, float* __restrict__ Asum) {
    __shared__ float sm[256];
    const _Float16* p = Azp + (size_t)blockIdx.x * PLANE;
    float s = 0.0f;
    for (int i = threadIdx.x; i < PLANE; i += 256) s += (float)p[i];
    sm[threadIdx.x] = s;
    __syncthreads();
    for (int off = 128; off > 0; off >>= 1) {
        if ((int)threadIdx.x < off) sm[threadIdx.x] += sm[threadIdx.x + off];
        __syncthreads();
    }
    if (threadIdx.x == 0) Asum[blockIdx.x] = sm[0] + 1.0f;
}

// ---------------------------------------------------------------------------
// Stage 3: argmax over N_MULT copies -> one-hot active mask (B,128).
// ---------------------------------------------------------------------------
__global__ void lcl_active(const float* __restrict__ Asum, float* __restrict__ act) {
    const int idx = blockIdx.x * 256 + threadIdx.x;
    if (idx >= 16 * 64) return;
    const int b = idx / 64, f = idx % 64;
    const bool first = (Asum[b * 128 + f] >= Asum[b * 128 + 64 + f]);  // argmax ties -> 0
    act[b * 128 + f]      = first ? 1.0f : 0.0f;
    act[b * 128 + 64 + f] = first ? 0.0f : 1.0f;
}

// ---------------------------------------------------------------------------
// Stage 4: Azpm = Azp * active  (pre-baked A_tgt; removes mask math from GEMM).
// ---------------------------------------------------------------------------
__global__ void lcl_build_azpm(const _Float16* __restrict__ Azp, const float* __restrict__ act,
                               _Float16* __restrict__ Azpm) {
    const int idx = blockIdx.x * 256 + threadIdx.x;
    const int c = (idx / PLANE) % 128;
    const int b = idx / (PLANE * 128);
    const _Float16 v = Azp[idx];
    Azpm[idx] = (act[b * 128 + c] > 0.5f) ? v : (_Float16)0.0f;
}

// ---------------------------------------------------------------------------
// Stage 5: correlation GEMM. 4 waves per block, batch dim split across waves
// (quarters the serial WMMA accumulation chain, 4x wave-level parallelism),
// LDS cross-wave reduction, coalesced stores into Kchg[tap][a][c].
// Kchg[t][a][c] = (1/B) * sum_{b,h,w} Azpm[b,a,h,w]*Azp[b,c,h+xo,w+yo];
// zero padding makes this exactly the reference's clipped-overlap sum.
// ---------------------------------------------------------------------------
__global__ void lcl_corr_wmma(const _Float16* __restrict__ Azpm, const _Float16* __restrict__ Azp,
                              const float* __restrict__ act, float* __restrict__ Kchg) {
    __shared__ float red[4][32][8];
    const int bid  = blockIdx.x;                  // 8*8*25 = 1600
    const int tap  = bid % 25;
    const int ct   = (bid / 25) % 8;
    const int at   = bid / 200;
    const int wv   = threadIdx.x >> 5;            // 0..3: batch-slice of this wave
    const int lane = threadIdx.x & 31;
    const int xo = tap / 5 - 2, yo = tap % 5 - 2;
    const int a0 = at * 16, c0 = ct * 16;
    const int arow = a0 + (lane & 15);
    const int ccol = c0 + (lane & 15);
    const int kbaseA = (lane >= 16) ? 8 : 0;      // A 16x32 f16 layout: K halves per lane group
    const int kbaseB = (lane >= 16) ? 8 : 0;      // B 32x16 f16 layout

    v8f acc = {};
    for (int bb = 0; bb < 4; ++bb) {
        const int b = wv + bb * 4;
        bool any = false;
        #pragma unroll
        for (int i = 0; i < 16; ++i) any |= (act[b * 128 + a0 + i] > 0.5f);
        if (!any) continue;                       // wave-uniform skip
        const _Float16* tb = Azpm + (size_t)(b * 128 + arow) * PLANE + 2 * PW + 8;
        const _Float16* sb = Azp  + (size_t)(b * 128 + ccol) * PLANE + (2 + xo) * PW + 8 + yo;
        for (int h = 0; h < 48; ++h) {
            const _Float16* tr = tb + h * PW;
            const _Float16* sr = sb + h * PW;
            #pragma unroll
            for (int wc = 0; wc < 2; ++wc) {
                const v8h t0 = *(const v8h*)(tr + wc * 32 + kbaseA);
                const v8h t1 = *(const v8h*)(tr + wc * 32 + kbaseA + 16);
                v16h bf;
                #pragma unroll
                for (int e = 0; e < 16; ++e) bf[e] = sr[wc * 32 + kbaseB + (e < 8 ? e : e + 8)];
                acc = __builtin_amdgcn_wmma_f32_16x16x32_f16(false, cat16(t0, t1), false, bf,
                                                             (short)0, acc, false, false);
            }
        }
    }
    #pragma unroll
    for (int r = 0; r < 8; ++r) red[wv][lane][r] = acc[r];
    __syncthreads();
    if (wv == 0) {
        #pragma unroll
        for (int r = 0; r < 8; ++r) {
            const float s = red[0][lane][r] + red[1][lane][r] + red[2][lane][r] + red[3][lane][r];
            const int a = a0 + r + ((lane >= 16) ? 8 : 0);
            Kchg[(size_t)tap * 16384 + a * 128 + ccol] = s * (1.0f / 16.0f);  // lanes contiguous in c
        }
    }
}

// ---------------------------------------------------------------------------
// Stage 6: W2[o][i][t] = minmax_t( 0.9*K[o][i][t] + 0.1*minmax_t(Kchg[:][i][o]) )
// stored f16 as [tap][o][i] for 16B-aligned contiguous A-fragment loads.
// ---------------------------------------------------------------------------
__global__ void lcl_wupd(const float* __restrict__ Kchg, const float* __restrict__ K,
                         _Float16* __restrict__ W2h) {
    const int idx = blockIdx.x * 256 + threadIdx.x;
    if (idx >= 128 * 128) return;
    const int o = idx / 128, i = idx % 128;
    float kc[25], mn = 3.4e38f, mx = -3.4e38f;
    #pragma unroll
    for (int t = 0; t < 25; ++t) {
        const float v = Kchg[(size_t)t * 16384 + i * 128 + o];   // K_change[a=i][c=o][t]
        kc[t] = v; mn = fminf(mn, v); mx = fmaxf(mx, v);
    }
    const float r1 = 1.0f / (mx - mn + EPS_);
    float wv[25], mn2 = 3.4e38f, mx2 = -3.4e38f;
    #pragma unroll
    for (int t = 0; t < 25; ++t) {
        const float u = (1.0f - ALPHA_) * K[((size_t)o * 128 + i) * 25 + t] + ALPHA_ * ((kc[t] - mn) * r1);
        wv[t] = u; mn2 = fminf(mn2, u); mx2 = fmaxf(mx2, u);
    }
    const float r2 = 1.0f / (mx2 - mn2 + EPS_);
    #pragma unroll
    for (int t = 0; t < 25; ++t)
        W2h[(size_t)t * 16384 + o * 128 + i] = (_Float16)((wv[t] - mn2) * r2);
}

// ---------------------------------------------------------------------------
// Stage 7: masked symmetric-padded activations, channel-last f16 (B,52,52,128).
// ---------------------------------------------------------------------------
__global__ void lcl_build_p2(const _Float16* __restrict__ Azp, const float* __restrict__ act,
                             _Float16* __restrict__ P2h) {
    const int idx = blockIdx.x * 256 + threadIdx.x;           // 16*52*52*128 = 5537792
    const int c = idx % 128;
    const int w = (idx / 128) % 52;
    const int h = (idx / (128 * 52)) % 52;
    const int b = idx / (128 * 52 * 52);
    const _Float16 v = Azp[(size_t)(b * 128 + c) * PLANE + (qmap(h) + 2) * PW + qmap(w) + 8];
    P2h[idx] = (act[b * 128 + c] > 0.5f) ? v : (_Float16)0.0f;
}

// ---------------------------------------------------------------------------
// Stage 8: conv2 as implicit GEMM via WMMA; fused gather + final output.
// One wave per (b, co_tile, h, w_tile); all fragment loads are b128.
// ---------------------------------------------------------------------------
__global__ void lcl_conv2_wmma(const _Float16* __restrict__ P2h, const _Float16* __restrict__ W2h,
                               const float* __restrict__ act, const float* __restrict__ A,
                               float* __restrict__ out) {
    const int wid  = blockIdx.x * 4 + (threadIdx.x >> 5);     // 16*8*48*3 = 18432 waves
    const int lane = threadIdx.x & 31;
    const int wt  = wid % 3;
    const int h   = (wid / 3) % 48;
    const int cot = (wid / 144) % 8;
    const int b   = wid / 1152;
    if (b >= 16) return;
    const int co0 = cot * 16;
    bool any = false;
    #pragma unroll
    for (int i = 0; i < 16; ++i) any |= (act[b * 128 + co0 + i] > 0.5f);
    if (!any) return;                                         // wave-uniform skip
    const int w0 = wt * 16;
    const int co = co0 + (lane & 15);
    const int n  = lane & 15;
    const int hiA = (lane >= 16) ? 8 : 0;
    const int hiB = (lane >= 16) ? 8 : 0;

    v8f acc = {};
    for (int tap = 0; tap < 25; ++tap) {
        const int xo = tap / 5, yo = tap % 5;
        const _Float16* xb = P2h + ((size_t)(b * 52 + h + xo) * 52 + (w0 + n + yo)) * 128 + hiB;
        const _Float16* wb = W2h + (size_t)tap * 16384 + co * 128 + hiA;
        #pragma unroll
        for (int kc = 0; kc < 4; ++kc) {
            const v8h a0v = *(const v8h*)(wb + kc * 32);
            const v8h a1v = *(const v8h*)(wb + kc * 32 + 16);
            const v8h b0v = *(const v8h*)(xb + kc * 32);
            const v8h b1v = *(const v8h*)(xb + kc * 32 + 16);
            acc = __builtin_amdgcn_wmma_f32_16x16x32_f16(false, cat16(a0v, a1v),
                                                         false, cat16(b0v, b1v),
                                                         (short)0, acc, false, false);
        }
    }
    #pragma unroll
    for (int r = 0; r < 8; ++r) {
        const int c = co0 + r + ((lane >= 16) ? 8 : 0);
        if (act[b * 128 + c] > 0.5f) {                        // exactly one active per (b, c&63)
            const size_t oi = (((size_t)b * 64 + (c & 63)) * 48 + h) * 48 + (w0 + n);
            out[oi] = A[oi] + THETA_ * (acc[r] * INV_FM);
        }
    }
}

// ---------------------------------------------------------------------------
extern "C" void kernel_launch(void* const* d_in, const int* in_sizes, int n_in,
                              void* d_out, int out_size, void* d_ws, size_t ws_size,
                              hipStream_t stream) {
    const float* A     = (const float*)d_in[0];   // (16,64,48,48)
    const float* K     = (const float*)d_in[1];   // (128,128,5,5)
    const float* noise = (const float*)d_in[2];   // (16,128,48,48)
    float* out = (float*)d_out;                   // (16,64,48,48)

    char* ws = (char*)d_ws;
    size_t off = 0;
    auto take = [&](size_t bytes) -> char* {
        char* p = ws + off;
        off = (off + bytes + 255) & ~(size_t)255;
        return p;
    };
    _Float16* Azp  = (_Float16*)take((size_t)16 * 128 * PLANE * 2);       // 17.0 MB
    _Float16* Azpm = (_Float16*)take((size_t)16 * 128 * PLANE * 2);       // 17.0 MB
    _Float16* P2h  = (_Float16*)take((size_t)16 * 52 * 52 * 128 * 2);     // 11.1 MB
    float*    Asum = (float*)   take((size_t)2048 * 4);
    float*    act  = (float*)   take((size_t)2048 * 4);
    float*    Kchg = (float*)   take((size_t)25 * 128 * 128 * 4);
    _Float16* W2h  = (_Float16*)take((size_t)25 * 128 * 128 * 2);

    lcl_build_azp <<<dim3(33280), dim3(256), 0, stream>>>(A, noise, Azp);
    lcl_asum      <<<dim3(2048),  dim3(256), 0, stream>>>(Azp, Asum);
    lcl_active    <<<dim3(4),     dim3(256), 0, stream>>>(Asum, act);
    lcl_build_azpm<<<dim3(33280), dim3(256), 0, stream>>>(Azp, act, Azpm);
    lcl_corr_wmma <<<dim3(1600),  dim3(128), 0, stream>>>(Azpm, Azp, act, Kchg);
    lcl_wupd      <<<dim3(64),    dim3(256), 0, stream>>>(Kchg, K, W2h);
    lcl_build_p2  <<<dim3(21632), dim3(256), 0, stream>>>(Azp, act, P2h);
    lcl_conv2_wmma<<<dim3(4608),  dim3(128), 0, stream>>>(P2h, W2h, act, A, out);

    (void)in_sizes; (void)n_in; (void)out_size; (void)ws_size;
}
